// RelativeAttention_88364657148484
// MI455X (gfx1250) — matrix operand order repaired and checked by reading, not compiled, not run on hardware
//
#include <hip/hip_runtime.h>
#include <hip/hip_bf16.h>

// ---------------------------------------------------------------------------
// RelativeAttention on gfx1250 (MI455X): all GEMMs via v_wmma_f32_16x16x32_bf16
// B=4, S=1024, D=1024, H=16, DK=64, BH=64
// Workspace layout (bytes):
//   [0,8M)     qb  bf16 [BH][S][DK]
//   [8M,16M)   kb  bf16 [BH][S][DK]
//   [16M,24M)  vb  bf16 [BH][S][DK]
//   [24M,32M)  xb  bf16 [B][S][H][DK]  (== [4096,1024] row-major attn output)
//   [32M,160M) s2  fp16 [BH][S(i)][S(j)]  (relative-bias scores)
// ---------------------------------------------------------------------------

#ifndef __has_builtin
#define __has_builtin(x) 0
#endif

typedef __attribute__((ext_vector_type(16))) __bf16 v16bf;
typedef __attribute__((ext_vector_type(8)))  float  v8f;

#if __has_builtin(__builtin_amdgcn_cvt_pk_bf16_f32)
typedef __attribute__((ext_vector_type(2))) __bf16 v2bf;
__device__ __forceinline__ unsigned pack2(float a, float b) {
  return __builtin_bit_cast(unsigned, __builtin_amdgcn_cvt_pk_bf16_f32(a, b));
}
__device__ __forceinline__ unsigned short f2bf(float f) {
  return (unsigned short)(pack2(f, 0.f) & 0xFFFFu);
}
#else
__device__ __forceinline__ unsigned short f2bf(float f) {
  unsigned u = __builtin_bit_cast(unsigned, f);
  u += 0x7FFFu + ((u >> 16) & 1u);  // round-to-nearest-even
  return (unsigned short)(u >> 16);
}
__device__ __forceinline__ unsigned pack2(float a, float b) {
  return (unsigned)f2bf(a) | ((unsigned)f2bf(b) << 16);
}
#endif

// A-fragment (16x32 bf16, MxK): lane holds row M=lane%16; element j has
// K = (j/8)*16 + (lane/16)*8 + j%8  -> two contiguous 16B loads.
__device__ __forceinline__ v16bf load_afrag(const unsigned* rowU, int lane) {
  union { uint4 q[2]; v16bf v; } f;
  int g4 = (lane >> 4) * 4;
  f.q[0] = *(const uint4*)(rowU + g4);
  f.q[1] = *(const uint4*)(rowU + 8 + g4);
  return f.v;
}
// B-fragment (32x16 bf16, KxN) from an [n][k] row-major tile.
__device__ __forceinline__ v16bf load_bfrag(const unsigned* rowU, int lane) {
  union { uint4 q[2]; v16bf v; } f;
  int g4 = (lane >> 4) * 4;
  f.q[0] = *(const uint4*)(rowU + g4);
  f.q[1] = *(const uint4*)(rowU + 8 + g4);
  return f.v;
}

__device__ __forceinline__ v8f wmma_bf16(v16bf a, v16bf b, v8f c) {
  return __builtin_amdgcn_wmma_f32_16x16x32_bf16(false, a, false, b, (short)0, c,
                                                 false, false);
}

// ---------------------------------------------------------------------------
// Tensor Data Mover: async 2D bf16 tile load into padded LDS rows.
// Pad: 4 DWORDs every 16 DWORDs -> 20-uint LDS row stride (bank-conflict free,
// 16B aligned). D# bitfields per CDNA5 ISA 8.3/8.4.
// ---------------------------------------------------------------------------
#if __has_builtin(__builtin_amdgcn_tensor_load_to_lds)
#define HAVE_TDM 1
typedef unsigned __attribute__((ext_vector_type(4))) u32x4;
typedef int __attribute__((ext_vector_type(4))) i32x4;
typedef int __attribute__((ext_vector_type(8))) i32x8;

__device__ __forceinline__ void tdm_load_2d_bf16(
    unsigned lds_off_bytes, const void* gaddr, unsigned tensor_d0,
    unsigned tensor_d1, unsigned long long stride0_elems, unsigned tile_d0,
    unsigned tile_d1) {
  unsigned long long ga = (unsigned long long)gaddr;
  u32x4 g0;
  g0[0] = 1u;                                   // count=1, user-mode
  g0[1] = lds_off_bytes;                        // lds_addr
  g0[2] = (unsigned)(ga & 0xFFFFFFFFu);         // global_addr[31:0]
  g0[3] = (unsigned)((ga >> 32) & 0x01FFFFFFu) | (2u << 30);  // type=2
  i32x8 g1;
  // wg_mask=0 | data_size=1(2B) | pad_enable | pad_interval=3(16dw) | pad_amount=3(4dw)
  g1[0] = (int)((1u << 16) | (1u << 20) | (3u << 22) | (3u << 25));
  g1[1] = (int)((tensor_d0 & 0xFFFFu) << 16);
  g1[2] = (int)((tensor_d0 >> 16) | ((tensor_d1 & 0xFFFFu) << 16));
  g1[3] = (int)((tensor_d1 >> 16) | (tile_d0 << 16));
  g1[4] = (int)(tile_d1 & 0xFFFFu);             // tile_dim1 | tile_dim2=0
  g1[5] = (int)(stride0_elems & 0xFFFFFFFFu);
  g1[6] = (int)((stride0_elems >> 32) & 0xFFFFu);
  g1[7] = 0;
  i32x4 z4 = {0, 0, 0, 0};
#if __clang_major__ >= 23
  i32x8 z8 = {0, 0, 0, 0, 0, 0, 0, 0};
  __builtin_amdgcn_tensor_load_to_lds(g0, g1, z4, z4, z8, 0);
#else
  __builtin_amdgcn_tensor_load_to_lds(g0, g1, z4, z4, 0);
#endif
}
#else
#define HAVE_TDM 0
#endif

#define LROW 20  // padded LDS row stride (uints) for 32-bf16 GEMM tile rows

// ---------------------------------------------------------------------------
// Projection GEMM: out[b,h,s,dk] = (X @ W + bias), X fp32 [4096,1024],
// W fp32 [1024,1024], out bf16 head-major [BH][S][DK].
// 256 thr = 8 waves; tile M=128 x N=128; wave = 32x64; K-step 32.
// ---------------------------------------------------------------------------
__global__ __launch_bounds__(256) void proj_kernel(
    const float* __restrict__ X, const float* __restrict__ W,
    const float* __restrict__ bias, unsigned short* __restrict__ out) {
  __shared__ unsigned lds_a[128 * LROW];  // [m][32 bf16] padded
  __shared__ unsigned lds_b[128 * LROW];  // [n][32 bf16] padded
  const int tid = threadIdx.x, lane = tid & 31, wave = tid >> 5;
  const int m0 = blockIdx.y * 128, n0 = blockIdx.x * 128;
  const int wm = wave & 3, wn = wave >> 2;

  v8f acc[2][4] = {};
  for (int k0 = 0; k0 < 1024; k0 += 32) {
    __syncthreads();
    {  // stage A (fp32 -> bf16)
      int row = tid >> 1, half = tid & 1;
      const float* src = X + (size_t)(m0 + row) * 1024 + k0 + half * 16;
      unsigned* dst = &lds_a[row * LROW + half * 8];
#pragma unroll
      for (int q = 0; q < 8; ++q) dst[q] = pack2(src[2 * q], src[2 * q + 1]);
    }
    {  // stage B transposed to [n][k]
      int n = tid & 127, ks = (tid >> 7) * 16;
      const float* src = W + (size_t)(k0 + ks) * 1024 + n0 + n;
      unsigned* dst = &lds_b[n * LROW + (ks >> 1)];
#pragma unroll
      for (int q = 0; q < 8; ++q)
        dst[q] = pack2(src[(2 * q) * 1024], src[(2 * q + 1) * 1024]);
    }
    __syncthreads();
    if (k0 + 32 < 1024) {
      __builtin_prefetch(X + (size_t)(m0 + (tid >> 1)) * 1024 + k0 + 32, 0, 1);
      __builtin_prefetch(W + (size_t)(k0 + 32 + (tid >> 7) * 16) * 1024 + n0 + (tid & 127), 0, 1);
    }
    v16bf a[2], b[4];
#pragma unroll
    for (int mi = 0; mi < 2; ++mi)
      a[mi] = load_afrag(&lds_a[(wm * 32 + mi * 16 + (lane & 15)) * LROW], lane);
#pragma unroll
    for (int ni = 0; ni < 4; ++ni)
      b[ni] = load_bfrag(&lds_b[(wn * 64 + ni * 16 + (lane & 15)) * LROW], lane);
#pragma unroll
    for (int mi = 0; mi < 2; ++mi)
#pragma unroll
      for (int ni = 0; ni < 4; ++ni) acc[mi][ni] = wmma_bf16(a[mi], b[ni], acc[mi][ni]);
  }
#pragma unroll
  for (int mi = 0; mi < 2; ++mi)
#pragma unroll
    for (int ni = 0; ni < 4; ++ni)
#pragma unroll
      for (int r = 0; r < 8; ++r) {
        int m = m0 + wm * 32 + mi * 16 + r + (lane >> 4) * 8;
        int n = n0 + wn * 64 + ni * 16 + (lane & 15);
        float v = acc[mi][ni][r] + bias[n];
        int b_ = m >> 10, s = m & 1023, h = n >> 6, dk = n & 63;
        out[((size_t)(b_ * 16 + h) * 1024 + s) * 64 + dk] = f2bf(v);
      }
}

// ---------------------------------------------------------------------------
// Relative-bias scores: for each query position i:
//   S2_i[64(bh), 1024(j)] = Q_i[64,64] @ rel[i]^T   (K = DK = 64), fp16 out.
// 256 thr = 8 waves; tile M=64 x N=128; wave = 16x64.
// ---------------------------------------------------------------------------
#define RROW 36  // padded stride (uints) for 64-bf16 rows
__global__ __launch_bounds__(256) void rel_kernel(
    const unsigned short* __restrict__ qb, const float* __restrict__ rel,
    _Float16* __restrict__ s2) {
  __shared__ unsigned lds_a[64 * RROW];   // Q_i [bh][64 bf16]
  __shared__ unsigned lds_b[128 * RROW];  // rel_i [j][64 bf16]
  const int i = blockIdx.y, j0 = blockIdx.x * 128;
  const int tid = threadIdx.x, lane = tid & 31, wave = tid >> 5;

  {  // A: gather Q rows (bh) at position i (already bf16)
    int row = tid >> 2, c = tid & 3;
    const unsigned* src = (const unsigned*)qb + ((size_t)row * 1024 + i) * 32 + c * 8;
    unsigned* dst = &lds_a[row * RROW + c * 8];
#pragma unroll
    for (int q = 0; q < 8; ++q) dst[q] = src[q];
  }
  {  // B: rel[i, j0+j, :] fp32 -> bf16 ([j][d] == [n][k])
    int j = tid >> 1, half = tid & 1;
    const float* src = rel + ((size_t)i * 1024 + j0 + j) * 64 + half * 32;
    unsigned* dst = &lds_b[j * RROW + half * 16];
#pragma unroll
    for (int q = 0; q < 16; ++q) dst[q] = pack2(src[2 * q], src[2 * q + 1]);
  }
  __syncthreads();

  const int mrow = (wave & 3) * 16, ncol = (wave >> 2) * 64;
  v8f acc[4] = {};
#pragma unroll
  for (int ks = 0; ks < 2; ++ks) {
    v16bf a = load_afrag(&lds_a[(mrow + (lane & 15)) * RROW + ks * 16], lane);
#pragma unroll
    for (int ni = 0; ni < 4; ++ni) {
      v16bf b = load_bfrag(&lds_b[(ncol + ni * 16 + (lane & 15)) * RROW + ks * 16], lane);
      acc[ni] = wmma_bf16(a, b, acc[ni]);
    }
  }
#pragma unroll
  for (int ni = 0; ni < 4; ++ni)
#pragma unroll
    for (int r = 0; r < 8; ++r) {
      int bh = mrow + r + (lane >> 4) * 8;
      int j = j0 + ncol + ni * 16 + (lane & 15);
      s2[((size_t)bh * 1024 + i) * 1024 + j] = (_Float16)acc[ni][r];
    }
}

// ---------------------------------------------------------------------------
// Fused attention per (bh, 16-row query tile). 128 thr = 4 waves.
// LDS rows padded: scores 1032 f32, pbuf/vT 516 uints -> conflict-free b128.
// ---------------------------------------------------------------------------
#define SCORES_F 66048   // 16*1032*4
#define PBUF_B 33024     // 16*516*4
#define VT_B 132096      // 64*516*4
#define ATTN_SMEM (SCORES_F + PBUF_B + VT_B + 512 + 128)
__global__ __launch_bounds__(128) void attn_kernel(
    const unsigned short* __restrict__ qb, const unsigned short* __restrict__ kb,
    const unsigned short* __restrict__ vb, const _Float16* __restrict__ s2,
    const int* __restrict__ mask, unsigned short* __restrict__ xb) {
  extern __shared__ __align__(16) char smem[];
  float* scores = (float*)smem;                          // [16][1032] f32
  unsigned* pbuf = (unsigned*)(smem + SCORES_F);         // [16][516] bf16x2
  unsigned* vT = (unsigned*)(smem + SCORES_F + PBUF_B);  // [64][516] bf16x2
  float* red = (float*)(smem + SCORES_F + PBUF_B + VT_B);  // [16][8]
  float* stat = red + 128;                               // [16] max, [16] 1/sum

  const int tid = threadIdx.x, lane = tid & 31, wave = tid >> 5;
  const int bh = blockIdx.y, b = bh >> 4, h = bh & 15;
  const int i0 = blockIdx.x * 16;
  const unsigned* qbu = (const unsigned*)qb;
  const unsigned* kbu = (const unsigned*)kb;
  const float sc = 0.125f;

  {  // stage V^T: vT[dk][j] <- vb[bh][j][dk]
    const unsigned* vsrc = (const unsigned*)vb + (size_t)bh * 1024 * 32;
    unsigned short* vTs = (unsigned short*)vT;
    for (int p = tid; p < 32768; p += 128) {
      int j = p >> 5, dkp = (p & 31) * 2;
      unsigned w = vsrc[p];
      vTs[dkp * 1032 + j] = (unsigned short)(w & 0xFFFFu);
      vTs[(dkp + 1) * 1032 + j] = (unsigned short)(w >> 16);
    }
  }

  v16bf qf0, qf1;  // Q fragments, K-dim 64 -> 2 frags, held in registers
  {
    const unsigned* qrow = qbu + ((size_t)bh * 1024 + i0 + (lane & 15)) * 32;
    qf0 = load_afrag(qrow, lane);
    qf1 = load_afrag(qrow + 16, lane);
  }

  // phase 1: scores = (q@k^T * sc + s2) * sc, mask, to LDS
  for (int jt = wave; jt < 64; jt += 4) {
    const unsigned* krow = kbu + ((size_t)bh * 1024 + jt * 16 + (lane & 15)) * 32;
    v16bf b0 = load_bfrag(krow, lane);
    v16bf b1 = load_bfrag(krow + 16, lane);
    v8f acc = {};
    acc = wmma_bf16(qf0, b0, acc);
    acc = wmma_bf16(qf1, b1, acc);
#pragma unroll
    for (int r = 0; r < 8; ++r) {
      int il = r + (lane >> 4) * 8;
      int i = i0 + il;
      int j = jt * 16 + (lane & 15);
      float v = (acc[r] * sc + (float)s2[((size_t)bh * 1024 + i) * 1024 + j]) * sc;
      if (mask[((size_t)b * 1024 + i) * 1024 + j] == 0) v = -1e9f;
      scores[il * 1032 + j] = v;
    }
  }
  __syncthreads();

  // phase 2: softmax per row; 8 consecutive threads per row (conflict-free)
  const int row = tid >> 3, grp = tid & 7;
  float mx = -3.4e38f;
  for (int j = grp; j < 1024; j += 8) mx = fmaxf(mx, scores[row * 1032 + j]);
  red[row * 8 + grp] = mx;
  __syncthreads();
  if (grp == 0) {
    float m = red[row * 8];
#pragma unroll
    for (int q = 1; q < 8; ++q) m = fmaxf(m, red[row * 8 + q]);
    stat[row] = m;
  }
  __syncthreads();
  const float m = stat[row];
  float sum = 0.f;
  for (int j = grp; j < 1024; j += 8) {
    float e = __expf(scores[row * 1032 + j] - m);
    scores[row * 1032 + j] = e;
    sum += e;
  }
  red[row * 8 + grp] = sum;
  __syncthreads();
  if (grp == 0) {
    float s = 0.f;
#pragma unroll
    for (int q = 0; q < 8; ++q) s += red[row * 8 + q];
    stat[16 + row] = 1.f / s;
  }
  __syncthreads();
  const float inv = stat[16 + row];
  for (int j = grp * 2; j < 1024; j += 16)
    pbuf[row * 516 + (j >> 1)] =
        pack2(scores[row * 1032 + j] * inv, scores[row * 1032 + j + 1] * inv);
  __syncthreads();

  // phase 3: out[16, dk-tile] = P[16,1024] @ V[1024,64]; wave -> 16-wide dk tile
  const int dkb = wave * 16;
  v8f oacc = {};
  for (int kt = 0; kt < 32; ++kt) {
    v16bf pa = load_afrag(&pbuf[(lane & 15) * 516 + kt * 16], lane);
    v16bf vf = load_bfrag(&vT[(dkb + (lane & 15)) * 516 + kt * 16], lane);
    oacc = wmma_bf16(pa, vf, oacc);
  }
#pragma unroll
  for (int r = 0; r < 8; ++r) {
    int i = i0 + r + (lane >> 4) * 8;
    int dk = dkb + (lane & 15);
    xb[((size_t)(b * 1024 + i) * 16 + h) * 64 + dk] = f2bf(oacc[r]);
  }
}

// ---------------------------------------------------------------------------
// Output projection: out f32 [4096,1024] = xb(bf16) @ Wo + bo.
// A tile staged by the Tensor Data Mover (async, padded rows) when available.
// ---------------------------------------------------------------------------
__global__ __launch_bounds__(256) void outproj_kernel(
    const unsigned short* __restrict__ Xb, const float* __restrict__ W,
    const float* __restrict__ bias, float* __restrict__ out) {
  __shared__ unsigned lds_a[128 * LROW];
  __shared__ unsigned lds_b[128 * LROW];
  const int tid = threadIdx.x, lane = tid & 31, wave = tid >> 5;
  const int m0 = blockIdx.y * 128, n0 = blockIdx.x * 128;
  const int wm = wave & 3, wn = wave >> 2;
  const unsigned* Xu = (const unsigned*)Xb;

  v8f acc[2][4] = {};
  for (int k0 = 0; k0 < 1024; k0 += 32) {
    __syncthreads();
#if HAVE_TDM
    if (tid == 0) {  // async TDM load of the 128x32 bf16 A tile into padded LDS
      tdm_load_2d_bf16((unsigned)(unsigned long long)(void*)lds_a,
                       Xb + (size_t)m0 * 1024 + k0,
                       /*tensor_d0=*/1024, /*tensor_d1=*/4096,
                       /*stride0=*/1024, /*tile_d0=*/32, /*tile_d1=*/128);
      __builtin_amdgcn_s_wait_tensorcnt(0);
    }
#else
    {
      int r2 = tid >> 1, half = tid & 1;
      const unsigned* src = Xu + (size_t)(m0 + r2) * 512 + (k0 >> 1) + half * 8;
      unsigned* dst = &lds_a[r2 * LROW + half * 8];
#pragma unroll
      for (int q = 0; q < 8; ++q) dst[q] = src[q];
    }
#endif
    {  // stage B transposed [n][k]
      int n = tid & 127, ks = (tid >> 7) * 16;
      const float* src = W + (size_t)(k0 + ks) * 1024 + n0 + n;
      unsigned* dst = &lds_b[n * LROW + (ks >> 1)];
#pragma unroll
      for (int q = 0; q < 8; ++q)
        dst[q] = pack2(src[(2 * q) * 1024], src[(2 * q + 1) * 1024]);
    }
    __syncthreads();
    if (k0 + 32 < 1024)
      __builtin_prefetch(W + (size_t)(k0 + 32 + (tid >> 7) * 16) * 1024 + n0 + (tid & 127), 0, 1);
    v16bf a[2], b[4];
#pragma unroll
    for (int mi = 0; mi < 2; ++mi)
      a[mi] = load_afrag(&lds_a[(wm * 32 + mi * 16 + (lane & 15)) * LROW], lane);
#pragma unroll
    for (int ni = 0; ni < 4; ++ni)
      b[ni] = load_bfrag(&lds_b[(wn * 64 + ni * 16 + (lane & 15)) * LROW], lane);
#pragma unroll
    for (int mi = 0; mi < 2; ++mi)
#pragma unroll
      for (int ni = 0; ni < 4; ++ni) acc[mi][ni] = wmma_bf16(a[mi], b[ni], acc[mi][ni]);
  }
#pragma unroll
  for (int mi = 0; mi < 2; ++mi)
#pragma unroll
    for (int ni = 0; ni < 4; ++ni)
#pragma unroll
      for (int r = 0; r < 8; ++r) {
        int mm = m0 + wm * 32 + mi * 16 + r + (lane >> 4) * 8;
        int nn = n0 + wn * 64 + ni * 16 + (lane & 15);
        out[(size_t)mm * 1024 + nn] = acc[mi][ni][r] + bias[nn];
      }
}

// ---------------------------------------------------------------------------
extern "C" void kernel_launch(void* const* d_in, const int* in_sizes, int n_in,
                              void* d_out, int out_size, void* d_ws, size_t ws_size,
                              hipStream_t stream) {
  (void)in_sizes; (void)n_in; (void)out_size; (void)ws_size;
  const float* query = (const float*)d_in[0];
  const float* key_  = (const float*)d_in[1];
  const float* value = (const float*)d_in[2];
  const int*   mask  = (const int*)d_in[3];
  const float* rel   = (const float*)d_in[4];
  const float* Wq = (const float*)d_in[5];
  const float* Wk = (const float*)d_in[6];
  const float* Wv = (const float*)d_in[7];
  const float* Wo = (const float*)d_in[8];
  const float* bq = (const float*)d_in[9];
  const float* bk = (const float*)d_in[10];
  const float* bv = (const float*)d_in[11];
  const float* bo = (const float*)d_in[12];
  float* out = (float*)d_out;

  char* ws = (char*)d_ws;
  unsigned short* qb = (unsigned short*)(ws);
  unsigned short* kb = (unsigned short*)(ws + ((size_t)8 << 20));
  unsigned short* vb = (unsigned short*)(ws + ((size_t)16 << 20));
  unsigned short* xb = (unsigned short*)(ws + ((size_t)24 << 20));
  _Float16* s2 = (_Float16*)(ws + ((size_t)32 << 20));

  dim3 gG(8, 32), bG(256);
  proj_kernel<<<gG, bG, 0, stream>>>(query, Wq, bq, qb);
  proj_kernel<<<gG, bG, 0, stream>>>(key_,  Wk, bk, kb);
  proj_kernel<<<gG, bG, 0, stream>>>(value, Wv, bv, vb);

  rel_kernel<<<dim3(8, 1024), 256, 0, stream>>>(qb, rel, s2);

  hipFuncSetAttribute((const void*)attn_kernel,
                      hipFuncAttributeMaxDynamicSharedMemorySize, ATTN_SMEM);
  attn_kernel<<<dim3(64, 64), 128, ATTN_SMEM, stream>>>(qb, kb, vb, s2, mask, xb);

  outproj_kernel<<<gG, bG, 0, stream>>>(xb, Wo, bo, out);
}
